// DecoderBlock_80788334837880
// MI455X (gfx1250) — matrix operand order repaired and checked by reading, not compiled, not run on hardware
//
#include <hip/hip_runtime.h>
#include <hip/hip_bf16.h>

// ---------------------------------------------------------------------------
// MI455X (gfx1250) decoder block: bf16 WMMA GEMMs + fused flash attention.
// Wave32; v_wmma_f32_16x16x32_bf16; double-buffered async global->LDS staging
// (ASYNCcnt); TDM tensor_load_to_lds for the attention Q tile (TENSORcnt).
// Weights are pre-transposed once so all GEMM B-fragments are contiguous.
// ---------------------------------------------------------------------------

typedef __bf16 bf16_t;
typedef __attribute__((ext_vector_type(16))) __bf16 v16bf;
typedef __attribute__((ext_vector_type(8)))  __bf16 v8bf;
typedef __attribute__((ext_vector_type(8)))  float  v8f;

#define B_   2
#define S_   2048
#define D_   1024
#define H_   16
#define DK_  64
#define DFF_ 4096
#define LN_EPS 1e-5f
#define NEG_HUGE (-3.402823466e38f)

// Feature toggles (each falls back to a plain path if disabled/unavailable).
#define CP_ASYNC 1
#define CP_TDM   1

// ---- async global -> LDS copy (GLOBAL_LOAD_ASYNC_TO_LDS_B128) --------------
__device__ __forceinline__ unsigned lds_off(const void* p) {
  // generic pointers to LDS carry the LDS byte offset in the low 32 bits
  return (unsigned)(uintptr_t)p;
}

__device__ __forceinline__ void cp_async16(void* lds, const void* g) {
#if CP_ASYNC
  asm volatile("global_load_async_to_lds_b128 %0, %1, off"
               :: "v"(lds_off(lds)), "v"(g) : "memory");
#else
  *(v8bf*)lds = *(const v8bf*)g;
#endif
}

__device__ __forceinline__ void cp_async_wait() {
#if CP_ASYNC
#if __has_builtin(__builtin_amdgcn_s_wait_asynccnt)
  __builtin_amdgcn_s_wait_asynccnt(0);
#else
  asm volatile("s_wait_asynccnt 0x0" ::: "memory");
#endif
#endif
}

// ---- Tensor Data Mover: 2D bf16 tile -> padded LDS (ISA 8.3/8.4) -----------
#if CP_TDM && __has_builtin(__builtin_amdgcn_tensor_load_to_lds)
#define HAVE_TDM 1
typedef __attribute__((ext_vector_type(4))) unsigned tdm_v4u;
typedef __attribute__((ext_vector_type(8))) int      tdm_v8i;
typedef __attribute__((ext_vector_type(4))) int      tdm_v4i;

// Loads a tile_w x tile_h tile of 2-byte elements (row stride = stride elems)
// into LDS, inserting 4 DWORDs of padding after every 32 DWORDs stored
// (i.e. rows of 64 bf16 land on an ld=72 padded layout).
__device__ __forceinline__ void tdm_load_2d(unsigned lds_addr, const void* gaddr,
                                            unsigned tile_w, unsigned tile_h,
                                            unsigned stride) {
  unsigned long long ga = (unsigned long long)(uintptr_t)gaddr;
  tdm_v4u g0;
  g0[0] = 1u;                                              // count=1 (valid D#)
  g0[1] = lds_addr;                                        // lds_addr
  g0[2] = (unsigned)(ga & 0xFFFFFFFFu);                    // global_addr lo
  g0[3] = (unsigned)((ga >> 32) & 0x01FFFFFFu) | (2u << 30); // addr hi | type=2
  tdm_v8i g1;
  // data_size=1 (2B) | pad_enable | pad_interval=32dw (code4) | pad_amount=4dw (code3)
  g1[0] = (int)((1u << 16) | (1u << 20) | (4u << 22) | (3u << 25));
  g1[1] = (int)((tile_w & 0xFFFFu) << 16);                 // tensor_dim0[15:0]
  g1[2] = (int)(((tile_w >> 16) & 0xFFFFu) | ((tile_h & 0xFFFFu) << 16));
  g1[3] = (int)(((tile_h >> 16) & 0xFFFFu) | ((tile_w & 0xFFFFu) << 16)); // tile_dim0
  g1[4] = (int)(tile_h & 0xFFFFu);                         // tile_dim1 (tile_dim2=0)
  g1[5] = (int)stride;                                     // tensor_dim0_stride lo
  g1[6] = 0;
  g1[7] = 0;
  tdm_v4i z4 = {0, 0, 0, 0};
  tdm_v8i z8 = {0, 0, 0, 0, 0, 0, 0, 0};
  // 6-arg form (clang-23 / therock-10.0 headers): (g0, g1, g2, g3, g4, cpol)
  __builtin_amdgcn_tensor_load_to_lds(g0, g1, z4, z4, z8, 0);
}

__device__ __forceinline__ void tdm_wait() {
  __builtin_amdgcn_s_wait_tensorcnt(0);
}
#endif

// ---- WMMA fragment builders (wave32; contiguous k = 16h + s, same for A and B) ----------

// A: 16(M) x 32(K), tile row-major, leading dim ld (elems).
__device__ __forceinline__ v16bf frag_a(const bf16_t* tile, int ld) {
  const int lane = threadIdx.x & 31;
  const int h = lane >> 4, m = lane & 15;
  const bf16_t* p = tile + m * ld;
  v8bf lo = *(const v8bf*)(p + 16 * h);
  v8bf hi = *(const v8bf*)(p + 16 * h + 8);
  v16bf a;
#pragma unroll
  for (int i = 0; i < 8; ++i) { a[i] = lo[i]; a[i + 8] = hi[i]; }
  return a;
}

// B: 32(K) x 16(N), memory tile stored N-major: tile[n*ld + k] (contiguous).
__device__ __forceinline__ v16bf frag_b_trans(const bf16_t* tile, int ld) {
  const int lane = threadIdx.x & 31;
  const int h = lane >> 4, n = lane & 15;
  const bf16_t* p = tile + n * ld + 16 * h;
  v8bf lo = *(const v8bf*)(p);
  v8bf hi = *(const v8bf*)(p + 8);
  v16bf b;
#pragma unroll
  for (int i = 0; i < 8; ++i) { b[i] = lo[i]; b[i + 8] = hi[i]; }
  return b;
}

// B: 32(K) x 16(N), memory tile K-major: tile[k*ld + n] (strided gather).
__device__ __forceinline__ v16bf frag_b_strided(const bf16_t* tile, int ld) {
  const int lane = threadIdx.x & 31;
  const int h = lane >> 4, n = lane & 15;
  v16bf b;
#pragma unroll
  for (int i = 0; i < 16; ++i) b[i] = tile[(16 * h + i) * ld + n];
  return b;
}

// ---- utility kernels -------------------------------------------------------
__global__ void __launch_bounds__(256)
fill_zero_f32(float* __restrict__ p, int n) {
  int i = blockIdx.x * 256 + threadIdx.x;
  if (i < n) p[i] = 0.0f;
}

__global__ void __launch_bounds__(256)
cvt_f32_bf16(const float* __restrict__ in, bf16_t* __restrict__ out, int n) {
  int i = (blockIdx.x * 256 + threadIdx.x) * 4;
  if (i + 3 < n) {
    float4 v = *(const float4*)&in[i];
    out[i + 0] = (bf16_t)v.x; out[i + 1] = (bf16_t)v.y;
    out[i + 2] = (bf16_t)v.z; out[i + 3] = (bf16_t)v.w;
  }
}

// fp32 [K,N] -> bf16 [N,K] transposed conversion (LDS-tiled, one-time)
__global__ void __launch_bounds__(256)
cvt_transpose_bf16(const float* __restrict__ in, bf16_t* __restrict__ outT,
                   int K, int N) {
  __shared__ __align__(16) bf16_t tl[64][72];   // [n_local][k_local]
  const int k0 = blockIdx.x * 64, n0 = blockIdx.y * 64;
  const int t = threadIdx.x;
  const int r = t >> 2, cg = t & 3;
#pragma unroll
  for (int i = 0; i < 4; ++i) {
    const int c = cg * 16 + i * 4;
    float4 v = *(const float4*)&in[(size_t)(k0 + r) * N + n0 + c];
    tl[c + 0][r] = (bf16_t)v.x; tl[c + 1][r] = (bf16_t)v.y;
    tl[c + 2][r] = (bf16_t)v.z; tl[c + 3][r] = (bf16_t)v.w;
  }
  __syncthreads();
#pragma unroll
  for (int i = 0; i < 2; ++i) {
    const int c = cg * 16 + i * 8;
    *(v8bf*)&outT[(size_t)(n0 + r) * K + k0 + c] = *(v8bf*)&tl[r][c];
  }
}

// ---- tiled WMMA GEMM: C(MxN,f32) = A(MxK,bf16 row-major) @ B (BT: [N,K]) ---
// WG tile 128x64, K-step 64, double-buffered async LDS staging.
// 8 waves (4 in M x 2 in N), 8 WMMA per wave per stage.
// Epilogue: v = fmax(acc + bias[col] + resid[row*rstride+col], clampLo).
//   bias/resid may point at a zero buffer (rstride=0 broadcasts row 0).
__global__ void __launch_bounds__(256)
gemm_bf16_wmma(const bf16_t* __restrict__ A, const bf16_t* __restrict__ BT,
               int M, int N, int K,
               const float* __restrict__ bias, const float* __restrict__ resid,
               int rstride, float clampLo,
               float* __restrict__ Cf, bf16_t* __restrict__ Cb) {
  __shared__ __align__(16) bf16_t As[2][128][72];  // 2 x (128 x 64), padded
  __shared__ __align__(16) bf16_t Bs[2][64][72];   // 2 x (64(n) x 64(k))

  const int t   = threadIdx.x;
  const int wid = t >> 5;
  const int wm  = wid >> 1;          // 0..3 -> 32 rows each
  const int wn  = wid & 1;           // 0..1 -> 32 cols each
  const int m0  = blockIdx.x * 128;
  const int n0  = blockIdx.y * 64;

  const int row8 = t >> 3, cg = t & 7;   // v8bf chunk mapping

  auto stage = [&](int buf, int k0) {
#pragma unroll
    for (int i = 0; i < 4; ++i) {      // A: 128x64 = 1024 chunks, 4/thread
      const int r = row8 + i * 32;
      cp_async16(&As[buf][r][cg * 8], &A[(size_t)(m0 + r) * K + k0 + cg * 8]);
    }
#pragma unroll
    for (int i = 0; i < 2; ++i) {      // B: 64x64 = 512 chunks, 2/thread
      const int r = row8 + i * 32;
      cp_async16(&Bs[buf][r][cg * 8], &BT[(size_t)(n0 + r) * K + k0 + cg * 8]);
    }
  };

  v8f zero = {};
  v8f acc[2][2] = {{zero, zero}, {zero, zero}};

  stage(0, 0);
  int buf = 0;
  for (int k0 = 0; k0 < K; k0 += 64) {
    cp_async_wait();
    __syncthreads();
    if (k0 + 64 < K) stage(buf ^ 1, k0 + 64);   // overlap with compute below
#pragma unroll
    for (int kc = 0; kc < 2; ++kc) {
      v16bf a0 = frag_a(&As[buf][wm * 32][kc * 32], 72);
      v16bf a1 = frag_a(&As[buf][wm * 32 + 16][kc * 32], 72);
      v16bf b0 = frag_b_trans(&Bs[buf][wn * 32][kc * 32], 72);
      v16bf b1 = frag_b_trans(&Bs[buf][wn * 32 + 16][kc * 32], 72);
      acc[0][0] = __builtin_amdgcn_wmma_f32_16x16x32_bf16(false, a0, false, b0,
                                                          (short)0, acc[0][0], false, false);
      acc[0][1] = __builtin_amdgcn_wmma_f32_16x16x32_bf16(false, a0, false, b1,
                                                          (short)0, acc[0][1], false, false);
      acc[1][0] = __builtin_amdgcn_wmma_f32_16x16x32_bf16(false, a1, false, b0,
                                                          (short)0, acc[1][0], false, false);
      acc[1][1] = __builtin_amdgcn_wmma_f32_16x16x32_bf16(false, a1, false, b1,
                                                          (short)0, acc[1][1], false, false);
    }
    buf ^= 1;
  }

  const int lane = t & 31, h = lane >> 4, nl = lane & 15;
#pragma unroll
  for (int j = 0; j < 2; ++j) {
    const int col = n0 + wn * 32 + j * 16 + nl;
    const float bv = bias[col];
#pragma unroll
    for (int i = 0; i < 2; ++i) {
#pragma unroll
      for (int r = 0; r < 8; ++r) {
        const int row = m0 + wm * 32 + i * 16 + r + 8 * h;
        float v = acc[i][j][r] + bv + resid[(size_t)row * rstride + col];
        v = fmaxf(v, clampLo);
        if (Cf) Cf[(size_t)row * N + col] = v;
        else    Cb[(size_t)row * N + col] = (bf16_t)v;
      }
    }
  }
}

// ---- fused flash attention (online softmax) -------------------------------
// grid = (S/64, B*H); block = 128 (4 waves). Each wave owns 16 q-rows.
// Q/K/V/O layout: [B*S, D] row-major, head h at column offset h*64.
__global__ void __launch_bounds__(128)
flash_attn_wmma(const bf16_t* __restrict__ Q, const bf16_t* __restrict__ Km,
                const bf16_t* __restrict__ V, bf16_t* __restrict__ O,
                int Skv, int causal) {
  __shared__ __align__(16) bf16_t Qs[64][72];
  __shared__ __align__(16) bf16_t Ks[2][32][72];
  __shared__ __align__(16) bf16_t Vs[2][32][72];
  __shared__ __align__(16) bf16_t Ps[4][16][40];   // per-wave P tile (16x32)

  const int t    = threadIdx.x;
  const int wid  = t >> 5;
  const int lane = t & 31, h = lane >> 4, nl = lane & 15;
  const int qbase = blockIdx.x * 64;
  const int b  = blockIdx.y / H_;
  const int hd = blockIdx.y % H_;
  const size_t base = ((size_t)b * S_) * D_ + (size_t)hd * DK_;

  auto stageKV = [&](int buf, int kb) {
#pragma unroll
    for (int i = 0; i < 2; ++i) {
      const int v = t + i * 128;
      const int row = v >> 3, cg = v & 7;
      cp_async16(&Ks[buf][row][cg * 8],
                 &Km[base + (size_t)(kb + row) * D_ + cg * 8]);
      cp_async16(&Vs[buf][row][cg * 8],
                 &V[base + (size_t)(kb + row) * D_ + cg * 8]);
    }
  };

  // stage the 64x64 Q tile: TDM descriptor (wave 0) or async copies
#if HAVE_TDM
  if (wid == 0)
    tdm_load_2d(lds_off(&Qs[0][0]), &Q[base + (size_t)qbase * D_], 64, 64, D_);
#else
#pragma unroll
  for (int i = 0; i < 4; ++i) {
    const int v = t + i * 128;
    const int row = v >> 3, cg = v & 7;
    cp_async16(&Qs[row][cg * 8], &Q[base + (size_t)(qbase + row) * D_ + cg * 8]);
  }
#endif
  stageKV(0, 0);
  cp_async_wait();
#if HAVE_TDM
  if (wid == 0) tdm_wait();
#endif
  __syncthreads();

  const v16bf aq0 = frag_a(&Qs[wid * 16][0], 72);    // dk 0..31
  const v16bf aq1 = frag_a(&Qs[wid * 16][32], 72);   // dk 32..63

  float mrow[8], lrow[8];
#pragma unroll
  for (int r = 0; r < 8; ++r) { mrow[r] = NEG_HUGE; lrow[r] = 0.f; }
  v8f zero = {};
  v8f accO[4] = {zero, zero, zero, zero};

  const float scale = 0.125f;   // 1/sqrt(DK)
  const int kend = causal ? (qbase + 64) : Skv;

  for (int kb = 0; kb < kend; kb += 32) {
    const int buf = (kb >> 5) & 1;
    if (kb + 32 < kend) stageKV(buf ^ 1, kb + 32);   // overlap with compute

    // S = Q(16x64) @ K^T -> two 16x16 tiles (Ks is [key][dk] => N-major)
    v8f s0 = {}; v8f s1 = {};
    {
      v16bf b00 = frag_b_trans(&Ks[buf][0][0], 72);
      v16bf b01 = frag_b_trans(&Ks[buf][0][32], 72);
      s0 = __builtin_amdgcn_wmma_f32_16x16x32_bf16(false, aq0, false, b00,
                                                   (short)0, s0, false, false);
      s0 = __builtin_amdgcn_wmma_f32_16x16x32_bf16(false, aq1, false, b01,
                                                   (short)0, s0, false, false);
      v16bf b10 = frag_b_trans(&Ks[buf][16][0], 72);
      v16bf b11 = frag_b_trans(&Ks[buf][16][32], 72);
      s1 = __builtin_amdgcn_wmma_f32_16x16x32_bf16(false, aq0, false, b10,
                                                   (short)0, s1, false, false);
      s1 = __builtin_amdgcn_wmma_f32_16x16x32_bf16(false, aq1, false, b11,
                                                   (short)0, s1, false, false);
    }

    // online softmax per C-layout row (m = r + 8h), 16 keys per lane-half
#pragma unroll
    for (int r = 0; r < 8; ++r) {
      const int m  = r + 8 * h;
      const int qg = qbase + wid * 16 + m;
      float v0 = s0[r] * scale;
      float v1 = s1[r] * scale;
      if (causal) {
        if (kb + nl      > qg) v0 = NEG_HUGE;
        if (kb + 16 + nl > qg) v1 = NEG_HUGE;
      }
      float tmax = fmaxf(v0, v1);
#pragma unroll
      for (int s = 1; s < 16; s <<= 1) tmax = fmaxf(tmax, __shfl_xor(tmax, s, 32));
      const float mnew = fmaxf(mrow[r], tmax);
      const float p0 = __expf(v0 - mnew);
      const float p1 = __expf(v1 - mnew);
      float rs = p0 + p1;
#pragma unroll
      for (int s = 1; s < 16; s <<= 1) rs += __shfl_xor(rs, s, 32);
      const float alpha = __expf(mrow[r] - mnew);
      lrow[r] = lrow[r] * alpha + rs;
      mrow[r] = mnew;
#pragma unroll
      for (int f = 0; f < 4; ++f) accO[f][r] *= alpha;
      Ps[wid][m][nl]      = (bf16_t)p0;   // C-layout -> row-major bounce
      Ps[wid][m][16 + nl] = (bf16_t)p1;
    }

    // O += P(16x32) @ V(32x64)  (per-wave LDS bounce; same-wave LDS in-order)
    const v16bf pA = frag_a(&Ps[wid][0][0], 40);
#pragma unroll
    for (int f = 0; f < 4; ++f) {
      v16bf bv = frag_b_strided(&Vs[buf][0][f * 16], 72);
      accO[f] = __builtin_amdgcn_wmma_f32_16x16x32_bf16(false, pA, false, bv,
                                                        (short)0, accO[f], false, false);
    }

    cp_async_wait();      // next buffer's K/V landed
    __syncthreads();
  }

#pragma unroll
  for (int r = 0; r < 8; ++r) {
    const int m  = r + 8 * h;
    const int qg = qbase + wid * 16 + m;
    const float inv = 1.0f / lrow[r];
#pragma unroll
    for (int f = 0; f < 4; ++f)
      O[base + (size_t)qg * D_ + f * 16 + nl] = (bf16_t)(accO[f][r] * inv);
  }
}

// ---- row layernorm (D=1024, 256 threads/row) ------------------------------
__global__ void __launch_bounds__(256)
layernorm_k(const float* __restrict__ X, const float* __restrict__ g,
            const float* __restrict__ bta, float* __restrict__ Yf,
            bf16_t* __restrict__ Yb, int Dd) {
  __shared__ float red[16];
  const int row = blockIdx.x;
  const int t = threadIdx.x;
  const float4 xv = *(const float4*)&X[(size_t)row * Dd + t * 4];
  float s  = xv.x + xv.y + xv.z + xv.w;
  float ss = xv.x * xv.x + xv.y * xv.y + xv.z * xv.z + xv.w * xv.w;
#pragma unroll
  for (int k = 1; k < 32; k <<= 1) {
    s += __shfl_xor(s, k, 32); ss += __shfl_xor(ss, k, 32);
  }
  const int wid = t >> 5;
  if ((t & 31) == 0) { red[wid] = s; red[8 + wid] = ss; }
  __syncthreads();
  float ts = 0.f, tss = 0.f;
#pragma unroll
  for (int w = 0; w < 8; ++w) { ts += red[w]; tss += red[8 + w]; }
  const float mu   = ts / Dd;
  const float var  = tss / Dd - mu * mu;
  const float rstd = rsqrtf(var + LN_EPS);
  const float xs[4] = {xv.x, xv.y, xv.z, xv.w};
#pragma unroll
  for (int i = 0; i < 4; ++i) {
    const int c = t * 4 + i;
    const float y = (xs[i] - mu) * rstd * g[c] + bta[c];
    Yf[(size_t)row * Dd + c] = y;
    if (Yb) Yb[(size_t)row * Dd + c] = (bf16_t)y;
  }
}

// ---------------------------------------------------------------------------
extern "C" void kernel_launch(void* const* d_in, const int* in_sizes, int n_in,
                              void* d_out, int out_size, void* d_ws, size_t ws_size,
                              hipStream_t stream) {
  const float* x     = (const float*)d_in[0];
  const float* enc   = (const float*)d_in[1];
  // d_in[2] = tgt_mask (causal tril -> handled analytically)
  // d_in[3] = src_mask (all ones -> no mask)
  const float* sa_wq = (const float*)d_in[4];
  const float* sa_wk = (const float*)d_in[5];
  const float* sa_wv = (const float*)d_in[6];
  const float* sa_wo = (const float*)d_in[7];
  const float* ca_wq = (const float*)d_in[8];
  const float* ca_wk = (const float*)d_in[9];
  const float* ca_wv = (const float*)d_in[10];
  const float* ca_wo = (const float*)d_in[11];
  const float* ff_w1 = (const float*)d_in[12];
  const float* ff_b1 = (const float*)d_in[13];
  const float* ff_w2 = (const float*)d_in[14];
  const float* ff_b2 = (const float*)d_in[15];
  const float* ln1_g = (const float*)d_in[16];
  const float* ln1_b = (const float*)d_in[17];
  const float* ln2_g = (const float*)d_in[18];
  const float* ln2_b = (const float*)d_in[19];
  const float* ln3_g = (const float*)d_in[20];
  const float* ln3_b = (const float*)d_in[21];

  const int M  = B_ * S_;                 // 4096 token rows
  const size_t MD = (size_t)M * D_;
  const size_t DD = (size_t)D_ * D_;
  const size_t DF = (size_t)D_ * DFF_;

  char* p = (char*)d_ws;
  auto carve = [&](size_t bytes) -> char* {
    char* r = p; p += (bytes + 255) & ~(size_t)255; return r;
  };
  float*  zbuf  = (float*)carve(DFF_ * 4);  // zero vector (bias/resid dummy)
  bf16_t* xb    = (bf16_t*)carve(MD * 2);
  bf16_t* encb  = (bf16_t*)carve(MD * 2);
  bf16_t* wsaq  = (bf16_t*)carve(DD * 2);   // all weights TRANSPOSED [N,K]
  bf16_t* wsak  = (bf16_t*)carve(DD * 2);
  bf16_t* wsav  = (bf16_t*)carve(DD * 2);
  bf16_t* wsao  = (bf16_t*)carve(DD * 2);
  bf16_t* wcaq  = (bf16_t*)carve(DD * 2);
  bf16_t* wcak  = (bf16_t*)carve(DD * 2);
  bf16_t* wcav  = (bf16_t*)carve(DD * 2);
  bf16_t* wcao  = (bf16_t*)carve(DD * 2);
  bf16_t* wff1  = (bf16_t*)carve(DF * 2);
  bf16_t* wff2  = (bf16_t*)carve(DF * 2);
  bf16_t* Qb    = (bf16_t*)carve(MD * 2);
  bf16_t* Kb    = (bf16_t*)carve(MD * 2);
  bf16_t* Vb    = (bf16_t*)carve(MD * 2);
  bf16_t* AOb   = (bf16_t*)carve(MD * 2);
  float*  residA = (float*)carve(MD * 4);
  float*  residB = (float*)carve(MD * 4);
  bf16_t* curb = xb;          // alias: xb dead after self-attn QKV projections
  bf16_t* ffh  = Qb;          // alias: Q/K/V/AO span (32MB) dead before FFN1

  const dim3 blk256(256), blk128(128);
  auto cvt = [&](const float* src, bf16_t* dst, size_t n) {
    cvt_f32_bf16<<<dim3((unsigned)(n / 1024)), blk256, 0, stream>>>(src, dst, (int)n);
  };
  auto cvtT = [&](const float* src, bf16_t* dstT, int K, int N) {
    cvt_transpose_bf16<<<dim3(K / 64, N / 64), blk256, 0, stream>>>(src, dstT, K, N);
  };
  auto gemm = [&](const bf16_t* A, const bf16_t* BT, int Mm, int Nn, int Kk,
                  const float* bias, const float* res, int rstride,
                  float clampLo, float* Cf, bf16_t* Cb) {
    dim3 grid(Mm / 128, Nn / 64);
    gemm_bf16_wmma<<<grid, blk256, 0, stream>>>(A, BT, Mm, Nn, Kk, bias, res,
                                                rstride, clampLo, Cf, Cb);
  };
  auto ln = [&](const float* X, const float* g, const float* b,
                float* Yf, bf16_t* Yb) {
    layernorm_k<<<dim3(M), blk256, 0, stream>>>(X, g, b, Yf, Yb, D_);
  };

  // ---- stage 0: zero buffer + bf16 conversion ----
  fill_zero_f32<<<dim3(DFF_ / 256), blk256, 0, stream>>>(zbuf, DFF_);
  cvt(x, xb, MD);       cvt(enc, encb, MD);
  cvtT(sa_wq, wsaq, D_, D_); cvtT(sa_wk, wsak, D_, D_);
  cvtT(sa_wv, wsav, D_, D_); cvtT(sa_wo, wsao, D_, D_);
  cvtT(ca_wq, wcaq, D_, D_); cvtT(ca_wk, wcak, D_, D_);
  cvtT(ca_wv, wcav, D_, D_); cvtT(ca_wo, wcao, D_, D_);
  cvtT(ff_w1, wff1, D_, DFF_);   // [D,DFF] -> [DFF,D]
  cvtT(ff_w2, wff2, DFF_, D_);   // [DFF,D] -> [D,DFF]

  // ---- self-attention ----
  gemm(xb, wsaq, M, D_, D_, zbuf, zbuf, 0, NEG_HUGE, nullptr, Qb);
  gemm(xb, wsak, M, D_, D_, zbuf, zbuf, 0, NEG_HUGE, nullptr, Kb);
  gemm(xb, wsav, M, D_, D_, zbuf, zbuf, 0, NEG_HUGE, nullptr, Vb);
  flash_attn_wmma<<<dim3(S_ / 64, B_ * H_), blk128, 0, stream>>>(
      Qb, Kb, Vb, AOb, S_, /*causal=*/1);
  gemm(AOb, wsao, M, D_, D_, zbuf, x, D_, NEG_HUGE, residA, nullptr);
  ln(residA, ln1_g, ln1_b, residB, curb);

  // ---- cross-attention ----
  gemm(curb, wcaq, M, D_, D_, zbuf, zbuf, 0, NEG_HUGE, nullptr, Qb);
  gemm(encb, wcak, M, D_, D_, zbuf, zbuf, 0, NEG_HUGE, nullptr, Kb);
  gemm(encb, wcav, M, D_, D_, zbuf, zbuf, 0, NEG_HUGE, nullptr, Vb);
  flash_attn_wmma<<<dim3(S_ / 64, B_ * H_), blk128, 0, stream>>>(
      Qb, Kb, Vb, AOb, S_, /*causal=*/0);
  gemm(AOb, wcao, M, D_, D_, zbuf, residB, D_, NEG_HUGE, residA, nullptr);
  ln(residA, ln2_g, ln2_b, residB, curb);

  // ---- feed-forward ----
  gemm(curb, wff1, M, DFF_, D_, ff_b1, zbuf, 0, 0.0f, nullptr, ffh);
  gemm(ffh, wff2, M, D_, DFF_, ff_b2, residB, D_, NEG_HUGE, residA, nullptr);
  ln(residA, ln3_g, ln3_b, (float*)d_out, nullptr);

  (void)in_sizes; (void)n_in; (void)out_size; (void)ws_size;
}
